// DBMAdvanced_58660663329204
// MI455X (gfx1250) — matrix operand order repaired and checked by reading, not compiled, not run on hardware
//
#include <hip/hip_runtime.h>

// ---------------------------------------------------------------------------
// DBM contrastive-divergence step for MI455X (gfx1250, wave32, WMMA).
// All heavy GEMMs run through one bf16 WMMA kernel (v_wmma_f32_16x16x32_bf16),
// fp32->bf16 conversion done while staging tiles into LDS. Large N=512,
// K=32768 GEMMs use split-K (grid.z) + a combine kernel for occupancy.
// ---------------------------------------------------------------------------

typedef __attribute__((ext_vector_type(16))) __bf16          v16bf;
typedef __attribute__((ext_vector_type(8)))  float           v8f;
typedef __attribute__((ext_vector_type(8)))  unsigned short  v8us;

union BFrag { v16bf v; v8us h[2]; };

#define DEVINL __device__ __forceinline__

static DEVINL unsigned short f32_to_bf16(float f) {
    unsigned u = __float_as_uint(f);
    unsigned r = u + 0x7FFFu + ((u >> 16) & 1u);   // round-to-nearest-even
    return (unsigned short)(r >> 16);
}

static DEVINL float urand01(unsigned seed, unsigned idx) {
    unsigned x = idx * 0x9E3779B9u ^ (seed * 0x85EBCA6Bu + 0x165667B1u);
    x ^= x >> 16; x *= 0x7FEB352Du;
    x ^= x >> 15; x *= 0x846CA68Bu;
    x ^= x >> 16;
    return (float)(x >> 8) * (1.0f / 16777216.0f);
}

static DEVINL float sigmoidf(float x) { return 1.0f / (1.0f + __expf(-x)); }

// Tile configuration: block = 256 threads = 8 wave32s arranged 2x4.
// Block tile 128x128, K-step 64; each wave owns a 64x32 sub-tile = 4x2 WMMAs.
constexpr int BM  = 128;
constexpr int BN  = 128;
constexpr int BK  = 64;
constexpr int LDW = BK + 8;   // +16B row pad -> conflict-free b128 LDS reads

enum { EP_SCALE = 0, EP_BIAS = 1, EP_SAMPLE = 2, EP_PARTIAL = 3 };

// Stage a 128xBK tile (generic strides, fp32 in global) into LDS as bf16.
static DEVINL void stage_tile(const float* __restrict__ G, long ms, long ks,
                              long r0, long k0, long kend,
                              unsigned short* __restrict__ lds, int tid) {
    if (ks == 1) {
        // contiguous along K: vectorized float4 path + prefetch of next tile
        #pragma unroll
        for (int i = 0; i < (BM * BK) / (4 * 256); ++i) {   // 8 iters
            int idx = tid + i * 256;
            int r   = idx / (BK / 4);
            int kq  = idx % (BK / 4);
            const float* gp = G + (r0 + r) * ms + (k0 + kq * 4);
            float4 f = *(const float4*)gp;
            if (k0 + BK < kend) __builtin_prefetch(gp + BK, 0, 3);
            unsigned short* d = lds + r * LDW + kq * 4;
            d[0] = f32_to_bf16(f.x); d[1] = f32_to_bf16(f.y);
            d[2] = f32_to_bf16(f.z); d[3] = f32_to_bf16(f.w);
        }
    } else {
        // generic strided path (transposed / odd-column views); row-inner for
        // coalescing when the row stride is 1 or 2.
        for (int i = tid; i < BM * BK; i += 256) {
            int k = i / BM;
            int r = i % BM;
            lds[r * LDW + k] = f32_to_bf16(G[(r0 + r) * ms + (k0 + k) * ks]);
        }
    }
}

// A fragment (16x32 bf16): lanes 0-15 -> row M=lane, K {0-7,16-23};
// lanes 16-31 -> row M=lane-16, K {8-15,24-31}. (ISA 7.12.2)
static DEVINL v16bf load_frag_a(const unsigned short* __restrict__ t,
                                int lane, int kk) {
    int half = (lane >> 4) & 1;
    int r    = lane & 15;
    const unsigned short* p = t + r * LDW + kk + half * 8;
    BFrag f;
    f.h[0] = *(const v8us*)p;
    f.h[1] = *(const v8us*)(p + 16);
    return f.v;
}

// B fragment (32x16 bf16), staged transposed in LDS as [n][k]:
// lanes 0-15 -> col n=lane, K {0-7,16-23}; lanes 16-31 -> col n, K {8-15,24-31}.
static DEVINL v16bf load_frag_b(const unsigned short* __restrict__ t,
                                int lane, int kk) {
    int grp = (lane >> 4) & 1;
    int n   = lane & 15;
    const unsigned short* p = t + n * LDW + kk + grp * 8;
    BFrag f;
    f.h[0] = *(const v8us*)p;
    f.h[1] = *(const v8us*)(p + 16);
    return f.v;
}

// C = op epilogue( A x B ), generic strides:
//   a(m,k) = A[m*a_ms + k*a_ks], b(k,n) = B[k*b_ks + n*b_ns]
//   out(m,n) = C[m*c_ms + n*c_ns]
// Split-K: blockIdx.z selects chunk [z*Ksplit, z*Ksplit+Ksplit) of K.
// EP_PARTIAL writes alpha*acc into partial slice z (C + z*M*c_ms ...).
__global__ __launch_bounds__(256)
void gemm_bf16_wmma(const float* __restrict__ A, long a_ms, long a_ks,
                    const float* __restrict__ B, long b_ks, long b_ns,
                    float* __restrict__ C, long c_ms, long c_ns,
                    const float* __restrict__ Cin, long ci_ms, long ci_ns,
                    const float* __restrict__ bias, int bias_stride,
                    int M, int N, int K, int Ksplit,
                    float alpha, float beta, int ep, unsigned seed) {
    __shared__ __align__(16) unsigned short lsA[BM * LDW];
    __shared__ __align__(16) unsigned short lsB[BN * LDW];

    const int  tid    = threadIdx.x;
    const int  lane   = tid & 31;
    const int  wave   = tid >> 5;
    const int  wm     = wave >> 2;            // 0..1
    const int  wn     = wave & 3;             // 0..3
    const long blockM = (long)blockIdx.y * BM;
    const long blockN = (long)blockIdx.x * BN;

    const long kstart = (long)blockIdx.z * Ksplit;
    long       kend   = kstart + Ksplit;
    if (kend > K) kend = K;

    v8f acc[4][2] = {};

    for (long k0 = kstart; k0 < kend; k0 += BK) {
        stage_tile(A, a_ms, a_ks, blockM, k0, kend, lsA, tid);
        stage_tile(B, b_ns, b_ks, blockN, k0, kend, lsB, tid);
        __syncthreads();
        #pragma unroll
        for (int kk = 0; kk < BK; kk += 32) {
            v16bf af[4], bf[2];
            #pragma unroll
            for (int ti = 0; ti < 4; ++ti)
                af[ti] = load_frag_a(lsA + (wm * 64 + ti * 16) * LDW, lane, kk);
            #pragma unroll
            for (int tj = 0; tj < 2; ++tj)
                bf[tj] = load_frag_b(lsB + (wn * 32 + tj * 16) * LDW, lane, kk);
            #pragma unroll
            for (int ti = 0; ti < 4; ++ti)
                #pragma unroll
                for (int tj = 0; tj < 2; ++tj)
                    acc[ti][tj] = __builtin_amdgcn_wmma_f32_16x16x32_bf16(
                        false, af[ti], false, bf[tj],
                        (short)0, acc[ti][tj], false, false);
        }
        __syncthreads();
    }

    // C layout: lane n = lane&15; VGPR r -> M = r + 8*(lane>>4).
    const int  nl   = lane & 15;
    const int  mb   = (lane >> 4) * 8;
    const long zoff = (long)blockIdx.z * (long)M * c_ms;
    #pragma unroll
    for (int ti = 0; ti < 4; ++ti)
    #pragma unroll
    for (int tj = 0; tj < 2; ++tj)
    #pragma unroll
    for (int r = 0; r < 8; ++r) {
        long m = blockM + wm * 64 + ti * 16 + mb + r;
        long n = blockN + wn * 32 + tj * 16 + nl;
        if (m < M && n < N) {
            float x = acc[ti][tj][r];
            if (ep == EP_PARTIAL) {
                C[zoff + m * c_ms + n * c_ns] = alpha * x;
            } else {
                float* op = C + m * c_ms + n * c_ns;
                if (ep == EP_SCALE) {
                    float o = alpha * x;
                    if (beta != 0.0f) o += beta * (*op);
                    *op = o;
                } else if (ep == EP_BIAS) {
                    *op = x + (bias ? bias[n * (long)bias_stride] : 0.0f);
                } else { // EP_SAMPLE: bernoulli(sigmoid(acc + Cin))
                    float s = x + (Cin ? Cin[m * ci_ms + n * ci_ns] : 0.0f);
                    float p = sigmoidf(s);
                    float u = urand01(seed, (unsigned)(m * (long)N + n));
                    *op = (u < p) ? 1.0f : 0.0f;
                }
            }
        }
    }
}

// out[i] = sum_z part[z*MN + i] + bias[i % N]   (split-K combine + bias)
__global__ __launch_bounds__(256)
void combine_bias_kernel(const float* __restrict__ part,
                         const float* __restrict__ bias, int bstr,
                         float* __restrict__ out, long MN, long N, int ks) {
    for (long i = blockIdx.x * (long)blockDim.x + threadIdx.x; i < MN;
         i += (long)gridDim.x * blockDim.x) {
        float s = 0.0f;
        for (int z = 0; z < ks; ++z) s += part[(long)z * MN + i];
        out[i] = s + bias[(i % N) * (long)bstr];
    }
}

__global__ void zero_kernel(float* __restrict__ p, long n) {
    for (long i = blockIdx.x * (long)blockDim.x + threadIdx.x; i < n;
         i += (long)gridDim.x * blockDim.x)
        p[i] = 0.0f;
}

// v_neg init: even cols = occ (v_data even cols), odd cols = random bits
// bern(u2 < u1) matching the clamp(uniform) semantics.
__global__ void init_vneg_kernel(const float* __restrict__ vd,
                                 float* __restrict__ vn, unsigned seed) {
    const long total = 2048L * 16384;
    for (long i = blockIdx.x * (long)blockDim.x + threadIdx.x; i < total;
         i += (long)gridDim.x * blockDim.x) {
        long m = i >> 14, j = i & 16383;
        long b = m * 32768 + 2 * j;
        vn[b] = vd[b];
        float p = urand01(seed, (unsigned)i);
        float u = urand01(seed ^ 0xA5A5A5A5u, (unsigned)i);
        vn[b + 1] = (u < p) ? 1.0f : 0.0f;
    }
}

// out[c*ostride+ooff] = scale * sum_r (P - Q)[r*rstride + c*cstride + coff]
__global__ __launch_bounds__(256)
void col_reduce_diff(const float* __restrict__ P, const float* __restrict__ Q,
                     float* __restrict__ out, int rows, long rstride,
                     long cstride, long coff, long ostride, long ooff,
                     float scale) {
    __shared__ float sm[256];
    long  c = blockIdx.x;
    float s = 0.0f;
    for (int r = threadIdx.x; r < rows; r += 256) {
        long idx = (long)r * rstride + c * cstride + coff;
        s += P[idx] - Q[idx];
    }
    sm[threadIdx.x] = s;
    __syncthreads();
    for (int off = 128; off > 0; off >>= 1) {
        if ((int)threadIdx.x < off) sm[threadIdx.x] += sm[threadIdx.x + off];
        __syncthreads();
    }
    if (threadIdx.x == 0) out[c * ostride + ooff] = sm[0] * scale;
}

__global__ __launch_bounds__(256)
void loss_partial_kernel(const float* __restrict__ vd,
                         const float* __restrict__ vn,
                         float* __restrict__ part) {
    __shared__ float sm[256];
    const long total = 2048L * 16384;
    float s = 0.0f;
    for (long i = blockIdx.x * 256L + threadIdx.x; i < total;
         i += (long)gridDim.x * 256) {
        long  m   = i >> 14, j = i & 16383;
        long  b   = m * 32768 + 2 * j;
        float occ = vd[b], st = vd[b + 1], sp = vn[b + 1];
        if (occ > 0.5f)
            s += -(st * __logf(sp + 1e-7f) +
                   (1.0f - st) * __logf(1.0f - sp + 1e-7f));
    }
    sm[threadIdx.x] = s;
    __syncthreads();
    for (int off = 128; off > 0; off >>= 1) {
        if ((int)threadIdx.x < off) sm[threadIdx.x] += sm[threadIdx.x + off];
        __syncthreads();
    }
    if (threadIdx.x == 0) part[blockIdx.x] = sm[0];
}

__global__ __launch_bounds__(256)
void loss_final_kernel(const float* __restrict__ part, float* __restrict__ out,
                       int n) {
    __shared__ float sm[256];
    float s = 0.0f;
    for (int i = threadIdx.x; i < n; i += 256) s += part[i];
    sm[threadIdx.x] = s;
    __syncthreads();
    for (int off = 128; off > 0; off >>= 1) {
        if ((int)threadIdx.x < off) sm[threadIdx.x] += sm[threadIdx.x + off];
        __syncthreads();
    }
    if (threadIdx.x == 0) out[0] = sm[0] / (2048.0f * 16384.0f);
}

extern "C" void kernel_launch(void* const* d_in, const int* in_sizes, int n_in,
                              void* d_out, int out_size, void* d_ws,
                              size_t ws_size, hipStream_t stream) {
    (void)in_sizes; (void)n_in; (void)ws_size;
    const float* v_data = (const float*)d_in[0];   // [2048, 32768]
    const float* W1     = (const float*)d_in[1];   // [512, 32768]
    const float* b_v    = (const float*)d_in[2];   // [32768]
    const float* b_h1   = (const float*)d_in[3];   // [512]
    const float* W2     = (const float*)d_in[4];   // [512, 512]
    const float* b_h2   = (const float*)d_in[5];   // [512]
    const float* U      = (const float*)d_in[6];   // [512, 32768]

    constexpr long Bz = 2048, NV = 32768, NH = 512, NS = 16384;
    constexpr int  KS = 4;                         // split-K factor for big GEMMs

    float* w    = (float*)d_ws;
    float* t1   = w;  w += Bz * NH;     // base/logit temp 1 (reused neg phase)
    float* t2   = w;  w += Bz * NH;     // base/logit temp 2
    float* h1d  = w;  w += Bz * NH;     // h1 positive samples
    float* h2d  = w;  w += Bz * NH;     // h2 positive samples
    float* h1n  = w;  w += Bz * NH;     // h1 negative samples
    float* h2n  = w;  w += Bz * NH;     // h2 negative samples
    float* vng  = w;  w += Bz * NV;     // v_neg (full layout)
    float* pvt  = w;  w += Bz * NS;     // prob_v odd-column logits
    float* prt  = w;  w += 1024;        // loss partials
    float* kpar = w;  w += (long)KS * Bz * NH;   // split-K partials

    float* out    = (float*)d_out;
    float* o_loss = out;
    float* o_dW1  = out + 1;
    float* o_dbv  = o_dW1 + NH * NV;
    float* o_dbh1 = o_dbv + NV;
    float* o_dW2  = o_dbh1 + NH;
    float* o_dbh2 = o_dW2 + NH * NH;
    float* o_dU   = o_dbh2 + NH;

    auto gemm = [&](const float* A, long ams, long aks,
                    const float* B, long bks, long bns,
                    float* C, long cms, long cns,
                    const float* Cin, long cims, long cins,
                    const float* bias, int bstr,
                    int M, int N, int K, float alpha, float beta,
                    int ep, unsigned seed) {
        dim3 g((N + BN - 1) / BN, (M + BM - 1) / BM, 1);
        gemm_bf16_wmma<<<g, 256, 0, stream>>>(A, ams, aks, B, bks, bns,
                                              C, cms, cns, Cin, cims, cins,
                                              bias, bstr, M, N, K, K,
                                              alpha, beta, ep, seed);
    };

    // Split-K big GEMM: C[2048,512] = A[2048,K] x B (both bf16-staged),
    // partials in kpar, then combine + bias into OutT.
    auto gemm_bigT = [&](const float* A, const float* B, float* OutT,
                         const float* bias) {
        dim3 g((NH + BN - 1) / BN, (Bz + BM - 1) / BM, KS);
        gemm_bf16_wmma<<<g, 256, 0, stream>>>(A, NV, 1, B, 1, NV,
                                              kpar, NH, 1,
                                              nullptr, 0, 0, nullptr, 0,
                                              (int)Bz, (int)NH, (int)NV,
                                              (int)(NV / KS),
                                              1.f, 0.f, EP_PARTIAL, 0);
        combine_bias_kernel<<<1024, 256, 0, stream>>>(kpar, bias, 1, OutT,
                                                      Bz * NH, NH, KS);
    };

    // ---- init -------------------------------------------------------------
    zero_kernel<<<256,  256, 0, stream>>>(h2d, Bz * NH);
    zero_kernel<<<256,  256, 0, stream>>>(h2n, Bz * NH);
    zero_kernel<<<2048, 256, 0, stream>>>(out, (long)out_size);
    init_vneg_kernel<<<2048, 256, 0, stream>>>(v_data, vng, 0xC0FFEEu);

    // ---- positive phase: loop-invariant big GEMMs hoisted -----------------
    // t1 = v_data @ W1.T + b_h1 ; t2 = v_data @ U.T + b_h2
    gemm_bigT(v_data, W1, t1, b_h1);
    gemm_bigT(v_data, U,  t2, b_h2);
    for (int it = 0; it < 5; ++it) {   // M_POS
        // h1 = bern(sig(t1 + h2 @ W2)); h2 = bern(sig(t2 + h1 @ W2.T))
        gemm(h2d, NH, 1, W2, NH, 1, h1d, NH, 1, t1, NH, 1, nullptr, 0,
             Bz, NH, NH, 1.f, 0.f, EP_SAMPLE, 100u + it);
        gemm(h1d, NH, 1, W2, 1, NH, h2d, NH, 1, t2, NH, 1, nullptr, 0,
             Bz, NH, NH, 1.f, 0.f, EP_SAMPLE, 200u + it);
    }

    // ---- negative phase (K_GIBBS = 1) -------------------------------------
    gemm_bigT(vng, W1, t1, b_h1);
    gemm(h2n, NH, 1, W2, NH, 1, h1n, NH, 1, t1, NH, 1, nullptr, 0,
         Bz, NH, NH, 1.f, 0.f, EP_SAMPLE, 300u);
    gemm_bigT(vng, U, t2, b_h2);
    gemm(h1n, NH, 1, W2, 1, NH, h2n, NH, 1, t2, NH, 1, nullptr, 0,
         Bz, NH, NH, 1.f, 0.f, EP_SAMPLE, 301u);
    // prob_v only at odd (sign) columns; even cols stay clamped to occ.
    gemm(h1n, NH, 1, W1 + 1, NV, 2, pvt, NS, 1, nullptr, 0, 0, b_v + 1, 2,
         Bz, NS, NH, 1.f, 0.f, EP_BIAS, 0);
    gemm(h2n, NH, 1, U + 1, NV, 2, vng + 1, NV, 2, pvt, NS, 1, nullptr, 0,
         Bz, NS, NH, 1.f, 0.f, EP_SAMPLE, 302u);

    // ---- final h1_neg / h2_neg with updated v_neg --------------------------
    gemm_bigT(vng, W1, t1, b_h1);
    gemm(h2n, NH, 1, W2, NH, 1, h1n, NH, 1, t1, NH, 1, nullptr, 0,
         Bz, NH, NH, 1.f, 0.f, EP_SAMPLE, 400u);
    gemm_bigT(vng, U, t2, b_h2);
    gemm(h1n, NH, 1, W2, 1, NH, h2n, NH, 1, t2, NH, 1, nullptr, 0,
         Bz, NH, NH, 1.f, 0.f, EP_SAMPLE, 401u);

    // ---- gradients ---------------------------------------------------------
    const float inv = 1.0f / 2048.0f;
    // dW1_full[:,1::2] = (h1_data^T v_odd - h1_neg^T vneg_odd)/B  (directly
    // transposed: C[512,16384] written with col stride 2 at offset 1)
    gemm(h1d, 1, NH, v_data + 1, NV, 2, o_dW1 + 1, NV, 2, nullptr, 0, 0,
         nullptr, 0, (int)NH, (int)NS, (int)Bz,  inv, 1.f, EP_SCALE, 0);
    gemm(h1n, 1, NH, vng + 1,    NV, 2, o_dW1 + 1, NV, 2, nullptr, 0, 0,
         nullptr, 0, (int)NH, (int)NS, (int)Bz, -inv, 1.f, EP_SCALE, 0);
    // dU_full[:,1::2]
    gemm(h2d, 1, NH, v_data + 1, NV, 2, o_dU + 1, NV, 2, nullptr, 0, 0,
         nullptr, 0, (int)NH, (int)NS, (int)Bz,  inv, 1.f, EP_SCALE, 0);
    gemm(h2n, 1, NH, vng + 1,    NV, 2, o_dU + 1, NV, 2, nullptr, 0, 0,
         nullptr, 0, (int)NH, (int)NS, (int)Bz, -inv, 1.f, EP_SCALE, 0);
    // dW2 = (h1_data^T h2_data - h1_neg^T h2_neg)/B
    gemm(h1d, 1, NH, h2d, NH, 1, o_dW2, NH, 1, nullptr, 0, 0,
         nullptr, 0, (int)NH, (int)NH, (int)Bz,  inv, 1.f, EP_SCALE, 0);
    gemm(h1n, 1, NH, h2n, NH, 1, o_dW2, NH, 1, nullptr, 0, 0,
         nullptr, 0, (int)NH, (int)NH, (int)Bz, -inv, 1.f, EP_SCALE, 0);

    col_reduce_diff<<<(int)NH, 256, 0, stream>>>(h1d, h1n, o_dbh1,
        (int)Bz, NH, 1, 0, 1, 0, inv);
    col_reduce_diff<<<(int)NH, 256, 0, stream>>>(h2d, h2n, o_dbh2,
        (int)Bz, NH, 1, 0, 1, 0, inv);
    col_reduce_diff<<<(int)NS, 256, 0, stream>>>(v_data, vng, o_dbv,
        (int)Bz, NV, 2, 1, 2, 1, inv);

    loss_partial_kernel<<<1024, 256, 0, stream>>>(v_data, vng, prt);
    loss_final_kernel<<<1, 256, 0, stream>>>(prt, o_loss, 1024);
}
